// LocalKNN_47485158425239
// MI455X (gfx1250) — matrix operand order repaired and checked by reading, not compiled, not run on hardware
//
#include <hip/hip_runtime.h>
#include <hip/hip_bf16.h>

typedef __attribute__((ext_vector_type(16))) _Float16 v16h;
typedef __attribute__((ext_vector_type(8)))  _Float16 v8h;
typedef __attribute__((ext_vector_type(8)))  float    v8f;

#define B_   64
#define WAY_ 5
#define D_   64
#define NQ_  1024
#define NS_  1024
#define EPS_ 1e-12f
#define INF_ __builtin_inff()

// ---------------------------------------------------------------------------
// Pass 0: zero the 320-float output (harness poisons it with 0xAA).
// ---------------------------------------------------------------------------
__global__ void knn_zero_out(float* out, int n) {
    int i = blockIdx.x * blockDim.x + threadIdx.x;
    if (i < n) out[i] = 0.0f;
}

// ---------------------------------------------------------------------------
// Pass 1a: query (B, D, Nq) fp32 -> qh[b][n][d] f16, L2-normalized along D.
// ---------------------------------------------------------------------------
__global__ __launch_bounds__(256) void knn_prep_q(const float* __restrict__ q,
                                                  _Float16* __restrict__ qh) {
    int idx = blockIdx.x * blockDim.x + threadIdx.x;   // b*NQ_ + n
    if (idx >= B_ * NQ_) return;
    int b = idx >> 10;
    int n = idx & (NQ_ - 1);
    const float* src = q + (size_t)b * D_ * NQ_ + n;   // stride NQ_ over d
    float ss = 0.0f;
#pragma unroll 8
    for (int d = 0; d < D_; ++d) {
        float v = src[(size_t)d * NQ_];
        ss += v * v;
    }
    float inv = 1.0f / fmaxf(__builtin_sqrtf(ss), EPS_);
    _Float16* dst = qh + (size_t)idx * D_;
#pragma unroll 8
    for (int d = 0; d < D_; ++d)
        dst[d] = (_Float16)(src[(size_t)d * NQ_] * inv);
}

// ---------------------------------------------------------------------------
// Pass 1b: support (B, Way, D, Ns) fp32 -> sh[bw][n][d] f16, normalized.
// ---------------------------------------------------------------------------
__global__ __launch_bounds__(256) void knn_prep_s(const float* __restrict__ s,
                                                  _Float16* __restrict__ sh) {
    int idx = blockIdx.x * blockDim.x + threadIdx.x;   // bw*NS_ + n
    if (idx >= B_ * WAY_ * NS_) return;
    int bw = idx >> 10;
    int n  = idx & (NS_ - 1);
    const float* src = s + (size_t)bw * D_ * NS_ + n;
    float ss = 0.0f;
#pragma unroll 8
    for (int d = 0; d < D_; ++d) {
        float v = src[(size_t)d * NS_];
        ss += v * v;
    }
    float inv = 1.0f / fmaxf(__builtin_sqrtf(ss), EPS_);
    _Float16* dst = sh + (size_t)idx * D_;
#pragma unroll 8
    for (int d = 0; d < D_; ++d)
        dst[d] = (_Float16)(src[(size_t)d * NS_] * inv);
}

// ---------------------------------------------------------------------------
// Pass 2: WMMA GEMM + running top-3 + reduction.
// Grid: (b*WAY + w) * 4 q-tiles of 256 rows; block = 256 threads = 8 waves.
// Each wave: 32 query rows (two M=16 A tiles), sweeps 64 N=16 chunks of Ns.
// Per chunk one B-fragment pair (K=64) feeds FOUR v_wmma_f32_16x16x32_f16.
// Ping-pong double buffer, distance-2 prefetch, no fragment copies.
// Top-3 insert is exactly 3 v_med3_f32 per value (max via med3(v,t0,+inf)).
// ---------------------------------------------------------------------------
__global__ __launch_bounds__(256) void knn_wmma(const _Float16* __restrict__ qh,
                                                const _Float16* __restrict__ sh,
                                                float* __restrict__ out) {
    const int wg    = blockIdx.x;
    const int qtile = wg & 3;          // 4 tiles of 256 query rows
    const int bw    = wg >> 2;         // b*WAY + w
    const int b     = bw / WAY_;
    const int wave  = (int)(threadIdx.x >> 5);
    const int lane  = (int)(threadIdx.x & 31);
    const int l16   = lane & 15;
    const int hi    = lane >> 4;       // 0: lanes 0-15, 1: lanes 16-31

    // ---- A fragments: rows rowbase..rowbase+31, two K=32 chunks each ----
    // 16-bit 16x32 A layout: lane l(<16) row M=l holds K=kb..kb+7 (VGPR0-3)
    // and K=kb+16..kb+23 (VGPR4-7), kb = (lane>=16) ? 8 : 0.
    const int rowbase = qtile * 256 + wave * 32;
    const int kb = hi * 8;
    const _Float16* qrow0 = qh + ((size_t)b * NQ_ + rowbase + l16) * D_;
    const _Float16* qrow1 = qrow0 + 16 * D_;

    v16h a0 = __builtin_shufflevector(*(const v8h*)(qrow0 + kb),
                                      *(const v8h*)(qrow0 + kb + 16),
                                      0,1,2,3,4,5,6,7,8,9,10,11,12,13,14,15);
    v16h a1 = __builtin_shufflevector(*(const v8h*)(qrow0 + 32 + kb),
                                      *(const v8h*)(qrow0 + 32 + kb + 16),
                                      0,1,2,3,4,5,6,7,8,9,10,11,12,13,14,15);
    v16h a2 = __builtin_shufflevector(*(const v8h*)(qrow1 + kb),
                                      *(const v8h*)(qrow1 + kb + 16),
                                      0,1,2,3,4,5,6,7,8,9,10,11,12,13,14,15);
    v16h a3 = __builtin_shufflevector(*(const v8h*)(qrow1 + 32 + kb),
                                      *(const v8h*)(qrow1 + 32 + kb + 16),
                                      0,1,2,3,4,5,6,7,8,9,10,11,12,13,14,15);

    // ---- running top-3 per C row-slot (sorted descending) ----
    // slots 0-7: rows rowbase + j + 8*hi ; slots 8-15: rows rowbase+16 + j + 8*hi
    float t0[16], t1[16], t2[16];
#pragma unroll
    for (int j = 0; j < 16; ++j) { t0[j] = -1e30f; t1[j] = -1e30f; t2[j] = -1e30f; }

    // sorted-triple insert: exactly 3 v_med3_f32, no canonicalize:
    //   t2' = med3(v, t1, t2); t1' = med3(v, t0, t1); t0' = med3(v, t0, +inf)
    auto insert3 = [&](int j, float v) {
        float n2 = __builtin_amdgcn_fmed3f(v, t1[j], t2[j]);
        float n1 = __builtin_amdgcn_fmed3f(v, t0[j], t1[j]);
        t0[j] = __builtin_amdgcn_fmed3f(v, t0[j], INF_);   // == max(v, t0)
        t1[j] = n1; t2[j] = n2;
    };

    // B layout (16-bit 32x16): lane l(<16) col N=l holds K=kb..kb+7 and
    // K=kb+16..kb+23, kb = (lane>=16) ? 8 : 0 (matches A).  16B loads.
    const _Float16* sbase = sh + ((size_t)bw << 16);   // bw * NS_ * D_
    const int koff = hi * 8;

    auto loadB = [&](int nc, v16h& b0, v16h& b1) {
        const _Float16* scol = sbase + (((size_t)(nc * 16 + l16)) << 6) + koff;
        b0 = __builtin_shufflevector(*(const v8h*)(scol), *(const v8h*)(scol + 16),
                                     0,1,2,3,4,5,6,7,8,9,10,11,12,13,14,15);        // K 0..31 slice
        b1 = __builtin_shufflevector(*(const v8h*)(scol + 32), *(const v8h*)(scol + 48),
                                     0,1,2,3,4,5,6,7,8,9,10,11,12,13,14,15);   // K 32..63 slice
    };

    auto compute = [&](const v16h& cb0, const v16h& cb1) {
        v8f cA = {}, cB = {};
        cA = __builtin_amdgcn_wmma_f32_16x16x32_f16(false, a0, false, cb0,
                                                    (short)0, cA, false, false);
        cB = __builtin_amdgcn_wmma_f32_16x16x32_f16(false, a2, false, cb0,
                                                    (short)0, cB, false, false);
        cA = __builtin_amdgcn_wmma_f32_16x16x32_f16(false, a1, false, cb1,
                                                    (short)0, cA, false, false);
        cB = __builtin_amdgcn_wmma_f32_16x16x32_f16(false, a3, false, cb1,
                                                    (short)0, cB, false, false);
#pragma unroll
        for (int j = 0; j < 8; ++j) insert3(j, cA[j]);
#pragma unroll
        for (int j = 0; j < 8; ++j) insert3(8 + j, cB[j]);
    };

    // ---- ping-pong pipelined sweep over the 64 N=16 chunks ----
    v16h b0a, b1a, b0b, b1b;
    loadB(0, b0a, b1a);
    loadB(1, b0b, b1b);
    for (int nc = 0; nc + 3 < NS_ / 16; nc += 2) {
        compute(b0a, b1a);            // consume buffer A (chunk nc)
        loadB(nc + 2, b0a, b1a);      // refill A (chunk nc+2)
        compute(b0b, b1b);            // consume buffer B (chunk nc+1)
        loadB(nc + 3, b0b, b1b);      // refill B (chunk nc+3)
    }
    compute(b0a, b1a);                // chunk 62
    compute(b0b, b1b);                // chunk 63

    // ---- merge top-3 across the 16 lanes sharing each row ----
    // XOR masks 1,2,4,8 stay within each 16-lane half (matches C layout).
#pragma unroll
    for (int m = 1; m < 16; m <<= 1) {
#pragma unroll
        for (int j = 0; j < 16; ++j) {
            float o0 = __shfl_xor(t0[j], m, 32);
            float o1 = __shfl_xor(t1[j], m, 32);
            float o2 = __shfl_xor(t2[j], m, 32);
            insert3(j, o0);
            insert3(j, o1);
            insert3(j, o2);
        }
    }

    // lanes 0 and 16 hold final top-3 for their 16 rows; sum and accumulate.
    if (l16 == 0) {
        float sum = 0.0f;
#pragma unroll
        for (int j = 0; j < 16; ++j) sum += t0[j] + t1[j] + t2[j];
        __hip_atomic_fetch_add(out + bw, sum, __ATOMIC_RELAXED,
                               __HIP_MEMORY_SCOPE_AGENT);
    }
}

// ---------------------------------------------------------------------------
extern "C" void kernel_launch(void* const* d_in, const int* in_sizes, int n_in,
                              void* d_out, int out_size, void* d_ws, size_t ws_size,
                              hipStream_t stream) {
    const float* q = (const float*)d_in[0];   // (64, 64, 32, 32)
    const float* s = (const float*)d_in[1];   // (64, 5, 64, 1024)
    float* out = (float*)d_out;               // (64, 5)

    // Workspace layout: qh f16 (8 MB), sh f16 (40 MB)
    _Float16* qh = (_Float16*)d_ws;
    _Float16* sh = (_Float16*)((char*)d_ws + (size_t)B_ * NQ_ * D_ * sizeof(_Float16));

    knn_zero_out<<<(B_ * WAY_ + 255) / 256, 256, 0, stream>>>(out, B_ * WAY_);

    knn_prep_q<<<(B_ * NQ_) / 256, 256, 0, stream>>>(q, qh);
    knn_prep_s<<<(B_ * WAY_ * NS_) / 256, 256, 0, stream>>>(s, sh);

    // (b*WAY + w) pairs * 4 q-tiles of 256 rows, 8 waves each.
    knn_wmma<<<B_ * WAY_ * (NQ_ / 256), 256, 0, stream>>>(qh, sh, out);
}
